// MultiHeadAttention_12043088298230
// MI455X (gfx1250) — compile-verified, matrix operand order checked by reading, not run on hardware
//
#include <hip/hip_runtime.h>
#include <hip/hip_bf16.h>

// ---------------------------------------------------------------------------
// MultiHeadAttention forward for MI455X (gfx1250, wave32, WMMA bf16 path).
//
// Compute-bound (~172 GFLOP vs ~130MB HBM traffic -> ~6us @ 23.3TB/s), so all
// matmuls go through v_wmma_f32_16x16x32_bf16 (K=32/issue vs K=4 for f32
// WMMA). fp32 inputs converted once to bf16 (RNE); accumulation stays fp32.
// Tile staging uses the CDNA5 async global->LDS path (ASYNCcnt) + prefetch.
// ---------------------------------------------------------------------------

typedef __bf16 bf16_t;
typedef __attribute__((ext_vector_type(16))) __bf16 v16bf;
typedef __attribute__((ext_vector_type(8)))  __bf16 v8bf;
typedef __attribute__((ext_vector_type(8)))  float  v8f;

static __device__ __forceinline__ bf16_t f2bf(float f) {
    unsigned u = __builtin_bit_cast(unsigned, f);
    unsigned r = u + 0x7FFFu + ((u >> 16) & 1u);   // round-to-nearest-even
    unsigned short h = (unsigned short)(r >> 16);
    return __builtin_bit_cast(bf16_t, h);
}

// Build a 16-element bf16 A/B fragment from two 16-byte LDS chunks.
// ISA 16-bit fragment layout: elems 0..7 = K kbase..kbase+7,
// elems 8..15 = K 16+kbase..16+kbase+7 (kbase = (lane>=16)*8).
static __device__ __forceinline__ v16bf load_frag(const bf16_t* p0, const bf16_t* p1) {
    v8bf a = *(const v8bf*)p0;
    v8bf b = *(const v8bf*)p1;
    return __builtin_shufflevector(a, b, 0,1,2,3,4,5,6,7,8,9,10,11,12,13,14,15);
}

// CDNA5 async direct global->LDS copy, 16 bytes per lane. Tracked by ASYNCcnt.
static __device__ __forceinline__ void async_copy_b128(void* lds_ptr, const void* gptr) {
    unsigned lds_off = (unsigned)(size_t)lds_ptr;           // LDS addr = addr[31:0]
    unsigned long long ga = (unsigned long long)(size_t)gptr;
    asm volatile("global_load_async_to_lds_b128 %0, %1, off"
                 :: "v"(lds_off), "v"(ga) : "memory");
}
static __device__ __forceinline__ void async_wait0() {
    asm volatile("s_wait_asynccnt 0x0" ::: "memory");
}

// ---------------------------------------------------------------------------
// fp32 -> bf16 conversion
// ---------------------------------------------------------------------------
__global__ __launch_bounds__(256)
void cvt_f32_bf16(const float* __restrict__ in, bf16_t* __restrict__ out, int n) {
    for (int i = blockIdx.x * blockDim.x + threadIdx.x; i < n;
         i += gridDim.x * blockDim.x)
        out[i] = f2bf(in[i]);
}

// ---------------------------------------------------------------------------
// GEMM: C[M,N] = A[M,K] * B[K,N] + bias[N]
//   A,B bf16; accum f32; output bf16 (projections) or f32 (final).
//   128x128 tile / workgroup (256 thr = 8 waves), wave = 32x64 sub-block.
// ---------------------------------------------------------------------------
#define GT_M 128
#define GT_N 128
#define GT_K 32
#define LDP  (GT_K + 8)   // LDS row stride (bf16 elems), 16B-aligned rows

template <bool OUT_F32>
__global__ __launch_bounds__(256)
void gemm_bf16(const bf16_t* __restrict__ A, const bf16_t* __restrict__ B,
               const float* __restrict__ bias, void* __restrict__ Cout,
               int M, int N, int K) {
    __shared__ bf16_t As[GT_M][LDP];   // [m][k]
    __shared__ bf16_t Bs[GT_N][LDP];   // [n][k]  (transposed on stage)

    const int tid   = threadIdx.x;
    const int lane  = tid & 31;
    const int wave  = tid >> 5;        // 0..7
    const int wm    = wave >> 1;       // 0..3 : 32-row block
    const int wn    = wave & 1;        // 0..1 : 64-col block
    const int l15   = lane & 15;
    const int lhalf = lane >> 4;
    const int kb    = lhalf * 8;

    const int m0 = blockIdx.y * GT_M;
    const int n0 = blockIdx.x * GT_N;

    const v8f vzero = {0.f,0.f,0.f,0.f,0.f,0.f,0.f,0.f};
    v8f acc[2][4];
#pragma unroll
    for (int s = 0; s < 2; ++s)
#pragma unroll
        for (int t = 0; t < 4; ++t) acc[s][t] = vzero;

    const int arow = tid >> 1;
    const int akc  = (tid & 1) * 16;
    const int bkk  = tid & 31;
    const int bnb  = (tid >> 5) * 16;

    for (int k0 = 0; k0 < K; k0 += GT_K) {
        // Stage A tile (128x32) via async direct-to-LDS: 2x 16B per thread.
        {
            const bf16_t* src = A + (size_t)(m0 + arow) * K + k0 + akc;
            async_copy_b128(&As[arow][akc],     src);
            async_copy_b128(&As[arow][akc + 8], src + 8);
        }
        // Stage B tile transposed: thread owns one k-row, 16 n-values.
        {
            const bf16_t* src = B + (size_t)(k0 + bkk) * N + n0 + bnb;
            v8bf b0 = *(const v8bf*)(src);
            v8bf b1 = *(const v8bf*)(src + 8);
#pragma unroll
            for (int i = 0; i < 8; ++i) {
                Bs[bnb + i][bkk]     = b0[i];
                Bs[bnb + 8 + i][bkk] = b1[i];
            }
        }
        // Prefetch next k-tile while this one is consumed.
        if (k0 + GT_K < K) {
            __builtin_prefetch(A + (size_t)(m0 + arow) * K + k0 + GT_K + akc, 0, 3);
            __builtin_prefetch(B + (size_t)(k0 + GT_K + bkk) * N + n0 + bnb, 0, 3);
        }
        async_wait0();
        __syncthreads();

        v16bf afrag[2];
#pragma unroll
        for (int s = 0; s < 2; ++s) {
            const bf16_t* ap = &As[wm * 32 + s * 16 + l15][0];
            afrag[s] = load_frag(ap + kb, ap + 16 + kb);
        }
#pragma unroll
        for (int t = 0; t < 4; ++t) {
            const bf16_t* bp = &Bs[wn * 64 + t * 16 + l15][0];
            v16bf bfrag = load_frag(bp + kb, bp + 16 + kb);
#pragma unroll
            for (int s = 0; s < 2; ++s)
                acc[s][t] = __builtin_amdgcn_wmma_f32_16x16x32_bf16(
                    false, afrag[s], false, bfrag, (short)0, acc[s][t],
                    false, false);
        }
        __syncthreads();
    }

    // Epilogue: bias + store. C layout: VGPR r -> M = r + 8*lhalf, lane -> N.
#pragma unroll
    for (int s = 0; s < 2; ++s)
#pragma unroll
        for (int t = 0; t < 4; ++t) {
            int col = n0 + wn * 64 + t * 16 + l15;
            float bv = bias[col];
#pragma unroll
            for (int r = 0; r < 8; ++r) {
                int row = m0 + wm * 32 + s * 16 + r + lhalf * 8;
                float v = acc[s][t][r] + bv;
                if (OUT_F32)
                    ((float*)Cout)[(size_t)row * N + col] = v;
                else
                    ((bf16_t*)Cout)[(size_t)row * N + col] = f2bf(v);
            }
        }
}

// ---------------------------------------------------------------------------
// Flash attention (causal). Q,K,V,Out: (b, n, d) bf16, head slice = h*128.
// Block = 64 query rows x one (b,h); 4 waves x 16 rows. Key tiles of 64.
// ---------------------------------------------------------------------------
#define AN  2048
#define AD  2048
#define ADK 128
#define KLD (ADK + 8)   // Ks row stride (16B-aligned rows)
#define VLD (64 + 8)    // Vt row stride (16B-aligned rows)

__global__ __launch_bounds__(128)
void flash_attn(const bf16_t* __restrict__ Q, const bf16_t* __restrict__ Km,
                const bf16_t* __restrict__ V, bf16_t* __restrict__ Out) {
    __shared__ bf16_t Ks[64][KLD];      // [key][dim]   natural: B-frag for QK^T
    __shared__ bf16_t Vt[ADK][VLD];     // [dim][key]   transposed: B-frag for PV
    __shared__ bf16_t Ps[4][16][64];    // per-wave P tile (C-layout -> A-layout)

    const int tid   = threadIdx.x;
    const int lane  = tid & 31;
    const int wave  = tid >> 5;   // 0..3
    const int l15   = lane & 15;
    const int lhalf = lane >> 4;
    const int kb    = lhalf * 8;

    const int q0 = blockIdx.x * 64;
    const int hh = blockIdx.y;
    const int bb = blockIdx.z;
    const size_t base = (size_t)bb * AN * AD + (size_t)hh * ADK;

    // Resident Q fragments: 16 rows x 128 dims = 4 fragments.
    v16bf qfrag[4];
    {
        const bf16_t* qp = Q + base + (size_t)(q0 + wave * 16 + l15) * AD;
#pragma unroll
        for (int c = 0; c < 4; ++c)
            qfrag[c] = load_frag(qp + c * 32 + kb, qp + c * 32 + 16 + kb);
    }

    const v8f vzero = {0.f,0.f,0.f,0.f,0.f,0.f,0.f,0.f};
    v8f o[8];
#pragma unroll
    for (int d = 0; d < 8; ++d) o[d] = vzero;
    float mrow[8], lrow[8];
#pragma unroll
    for (int r = 0; r < 8; ++r) { mrow[r] = -3.0e38f; lrow[r] = 0.f; }

    const float scale = 0.08838834764831845f;  // 1/sqrt(128)
    const int myrow0 = q0 + wave * 16 + lhalf * 8;   // row of accum element r

    const int srow = tid >> 1;            // staging: 2 threads per key row
    const int scc  = (tid & 1) * 64;

    const int jtiles = blockIdx.x + 1;    // causal: jt <= qt
    for (int jt = 0; jt < jtiles; ++jt) {
        const int k0g = jt * 64;
        const bf16_t* ksrc = Km + base + (size_t)(k0g + srow) * AD + scc;
        const bf16_t* vsrc = V  + base + (size_t)(k0g + srow) * AD + scc;

        // K tile: async direct-to-LDS (natural layout), 8x 16B per thread.
#pragma unroll
        for (int i = 0; i < 64; i += 8)
            async_copy_b128(&Ks[srow][scc + i], ksrc + i);

        // V tile: load to VGPRs, store transposed (B-frag for PV wants [dim][key]).
        {
            v8bf vv[8];
#pragma unroll
            for (int j = 0; j < 8; ++j) vv[j] = *(const v8bf*)(vsrc + 8 * j);
#pragma unroll
            for (int j = 0; j < 8; ++j)
#pragma unroll
                for (int e = 0; e < 8; ++e)
                    Vt[scc + 8 * j + e][srow] = vv[j][e];
        }
        // Prefetch next tile's K/V rows while computing on this one.
        if (jt + 1 < jtiles) {
            __builtin_prefetch(ksrc + (size_t)64 * AD, 0, 3);
            __builtin_prefetch(vsrc + (size_t)64 * AD, 0, 3);
        }
        async_wait0();
        __syncthreads();

        // S = Q K^T : 4 subtiles (16 keys each), K-dim 128 in 4 chunks.
        v8f s[4];
#pragma unroll
        for (int t = 0; t < 4; ++t) {
            s[t] = vzero;
            const bf16_t* kp = &Ks[t * 16 + l15][0];
#pragma unroll
            for (int c = 0; c < 4; ++c) {
                v16bf kf = load_frag(kp + c * 32 + kb, kp + c * 32 + 16 + kb);
                s[t] = __builtin_amdgcn_wmma_f32_16x16x32_bf16(
                    false, qfrag[c], false, kf, (short)0, s[t], false, false);
            }
        }

        // Scale + causal mask.
#pragma unroll
        for (int t = 0; t < 4; ++t) {
            int colg = k0g + t * 16 + l15;
#pragma unroll
            for (int r = 0; r < 8; ++r) {
                float sv = s[t][r] * scale;
                s[t][r] = (colg <= myrow0 + r) ? sv : -3.0e38f;
            }
        }

        // Online softmax: row-max / row-sum across the 16-lane half.
        float newm[8], corr[8], rs[8];
#pragma unroll
        for (int r = 0; r < 8; ++r) {
            float mx = fmaxf(fmaxf(s[0][r], s[1][r]), fmaxf(s[2][r], s[3][r]));
#pragma unroll
            for (int msk = 1; msk < 16; msk <<= 1)
                mx = fmaxf(mx, __shfl_xor(mx, msk, 32));
            newm[r] = fmaxf(mrow[r], mx);
            corr[r] = __expf(mrow[r] - newm[r]);
            mrow[r] = newm[r];
            lrow[r] *= corr[r];
            rs[r]    = 0.f;
        }
#pragma unroll
        for (int t = 0; t < 4; ++t)
#pragma unroll
            for (int r = 0; r < 8; ++r) {
                float p = __expf(s[t][r] - newm[r]);
                s[t][r] = p;
                rs[r] += p;
            }
#pragma unroll
        for (int r = 0; r < 8; ++r) {
            float sum = rs[r];
#pragma unroll
            for (int msk = 1; msk < 16; msk <<= 1)
                sum += __shfl_xor(sum, msk, 32);
            lrow[r] += sum;
        }
#pragma unroll
        for (int d = 0; d < 8; ++d)
#pragma unroll
            for (int r = 0; r < 8; ++r) o[d][r] *= corr[r];

        // P (C-layout) -> LDS -> A-layout fragments.
#pragma unroll
        for (int t = 0; t < 4; ++t)
#pragma unroll
            for (int r = 0; r < 8; ++r)
                Ps[wave][r + lhalf * 8][t * 16 + l15] = f2bf(s[t][r]);
        __syncthreads();

        // O += P @ V : key dim 64 in 2 chunks, 8 dk subtiles.
#pragma unroll
        for (int c = 0; c < 2; ++c) {
            const bf16_t* pp = &Ps[wave][l15][c * 32];
            v16bf pf = load_frag(pp + kb, pp + 16 + kb);
#pragma unroll
            for (int d = 0; d < 8; ++d) {
                const bf16_t* vp = &Vt[d * 16 + l15][c * 32];
                v16bf vf = load_frag(vp + kb, vp + 16 + kb);
                o[d] = __builtin_amdgcn_wmma_f32_16x16x32_bf16(
                    false, pf, false, vf, (short)0, o[d], false, false);
            }
        }
        __syncthreads();   // protect Ks/Vt/Ps before next tile
    }

    // Normalize + store (b,n,d) bf16.
#pragma unroll
    for (int r = 0; r < 8; ++r) {
        float inv = 1.0f / lrow[r];
        bf16_t* op = Out + base + (size_t)(myrow0 + r) * AD;
#pragma unroll
        for (int d = 0; d < 8; ++d)
            op[d * 16 + l15] = f2bf(o[d][r] * inv);
    }
}

// ---------------------------------------------------------------------------
// Launch: cvt -> 3x QKV GEMM -> flash attention -> output-projection GEMM.
// ---------------------------------------------------------------------------
extern "C" void kernel_launch(void* const* d_in, const int* in_sizes, int n_in,
                              void* d_out, int out_size, void* d_ws, size_t ws_size,
                              hipStream_t stream) {
    (void)in_sizes; (void)n_in; (void)out_size; (void)ws_size;

    const float* x  = (const float*)d_in[0];
    const float* Wq = (const float*)d_in[1];
    const float* bq = (const float*)d_in[2];
    const float* Wk = (const float*)d_in[3];
    const float* bk = (const float*)d_in[4];
    const float* Wv = (const float*)d_in[5];
    const float* bv = (const float*)d_in[6];
    const float* Wo = (const float*)d_in[7];
    const float* bo = (const float*)d_in[8];

    constexpr int    B  = 2, N = 2048, D = 2048;
    constexpr size_t MD = (size_t)B * N * D;   // 8M elems
    constexpr size_t WD = (size_t)D * D;       // 4M elems

    bf16_t* ws  = (bf16_t*)d_ws;
    bf16_t* xb  = ws;
    bf16_t* wqb = ws + MD;
    bf16_t* wkb = wqb + WD;
    bf16_t* wvb = wkb + WD;
    bf16_t* wob = wvb + WD;
    bf16_t* Qb  = wob + WD;
    bf16_t* Kb  = Qb + MD;
    bf16_t* Vb  = Kb + MD;
    bf16_t* Ab  = Vb + MD;       // attention output (b,n,d) bf16

    cvt_f32_bf16<<<2048, 256, 0, stream>>>(x,  xb,  (int)MD);
    cvt_f32_bf16<<<1024, 256, 0, stream>>>(Wq, wqb, (int)WD);
    cvt_f32_bf16<<<1024, 256, 0, stream>>>(Wk, wkb, (int)WD);
    cvt_f32_bf16<<<1024, 256, 0, stream>>>(Wv, wvb, (int)WD);
    cvt_f32_bf16<<<1024, 256, 0, stream>>>(Wo, wob, (int)WD);

    dim3 gg(D / GT_N, (B * N) / GT_M);   // 16 x 32
    gemm_bf16<false><<<gg, 256, 0, stream>>>(xb, wqb, bq, Qb, B * N, D, D);
    gemm_bf16<false><<<gg, 256, 0, stream>>>(xb, wkb, bk, Kb, B * N, D, D);
    gemm_bf16<false><<<gg, 256, 0, stream>>>(xb, wvb, bv, Vb, B * N, D, D);

    dim3 ag(N / 64, 16, B);              // 32 x 16 x 2
    flash_attn<<<ag, 128, 0, stream>>>(Qb, Kb, Vb, Ab);

    gemm_bf16<true><<<gg, 256, 0, stream>>>(Ab, wob, bo, (float*)d_out, B * N, D, D);
}
